// STM_46978352283912
// MI455X (gfx1250) — matrix operand order repaired and checked by reading, not compiled, not run on hardware
//
#include <hip/hip_runtime.h>
#include <hip/hip_bf16.h>

typedef __attribute__((ext_vector_type(16))) _Float16 v16h;
typedef __attribute__((ext_vector_type(8)))  _Float16 v8h;
typedef __attribute__((ext_vector_type(8)))  float    v8f;
typedef __attribute__((ext_vector_type(4)))  int      v4i;

#define MTOK 32768                // B*H*2W tokens
#define MIMG 16384                // B*H*W tokens per de-interleaved image

__device__ __forceinline__ v8f wmma16(v16h a, v16h b, v8f c) {
  return __builtin_amdgcn_wmma_f32_16x16x32_f16(
      /*neg_a=*/false, a, /*neg_b=*/false, b,
      /*c_mod=*/(short)0, c, /*reuse_a=*/false, /*reuse_b=*/false);
}

// A-fragment: 16x32 f16, row-major source. Lane<16: K={0..7,16..23}; lane>=16: +8.
__device__ __forceinline__ v16h load_a_frag(const _Float16* __restrict__ Arow,
                                            int kbase, int hs) {
  const _Float16* p = Arow + kbase + hs * 8;
  v8h lo = *(const v8h*)(p);
  v8h hi = *(const v8h*)(p + 16);
  v16h a;
#pragma unroll
  for (int i = 0; i < 8; ++i) { a[i] = lo[i]; a[8 + i] = hi[i]; }
  return a;
}

// ---- async global->LDS staging (CDNA5 ASYNCcnt path, with fallbacks) ------
#if __has_builtin(__builtin_amdgcn_global_load_async_to_lds_b128)
#define ASYNC_BUILTIN 1
#endif

__device__ __forceinline__ void stage16(const void* g, void* l) {
#if defined(ASYNC_BUILTIN)
  __builtin_amdgcn_global_load_async_to_lds_b128(
      (__attribute__((address_space(1))) v4i*)g,
      (__attribute__((address_space(3))) v4i*)l, 0, 0);
#elif defined(__gfx1250__)
  unsigned loff = (unsigned)(size_t)(__attribute__((address_space(3))) void*)l;
  unsigned long long ga = (unsigned long long)(size_t)g;
  asm volatile("global_load_async_to_lds_b128 %0, %1, off"
               :: "v"(loff), "v"(ga) : "memory");
#else
  *(v8h*)l = *(const v8h*)g;
#endif
}

__device__ __forceinline__ void stage_wait() {
#if defined(__gfx1250__)
#if __has_builtin(__builtin_amdgcn_s_wait_asynccnt)
  __builtin_amdgcn_s_wait_asynccnt(0);
#else
  asm volatile("s_wait_asynccnt 0x0" ::: "memory");
#endif
#endif
}

// ---------------- weight packing into per-lane B fragments -----------------
// dst idx = ((kc*nTiles + nt)*32 + lane)*16 + e ; K = kc*32 + (lane<16?0:8) + (e&7) + (e>>3)*16 ;
// n = nt*16 + (lane&15). conv==0: src is (Nd,Kd) row-major (out-major weights).
// conv==1: src is rb_w (o,i,3,3); K = tap*128 + i.
__global__ void pack_b_kernel(const float* __restrict__ src, _Float16* __restrict__ dst,
                              int Kd, int Nd, int conv) {
  int idx = blockIdx.x * blockDim.x + threadIdx.x;
  if (idx >= Kd * Nd) return;
  int e = idx & 15;
  int lane = (idx >> 4) & 31;
  int fi = idx >> 9;
  int nTiles = Nd >> 4;
  int kc = fi / nTiles, nt = fi % nTiles;
  int K = (kc << 5) + ((lane >> 4) << 3) + (e & 7) + ((e >> 3) << 4);
  int n = (nt << 4) + (lane & 15);
  float v;
  if (conv) { int i = K & 127, t = K >> 7; v = src[((size_t)n * 128 + i) * 9 + t]; }
  else      { v = src[(size_t)n * Kd + K]; }
  dst[idx] = (_Float16)v;
}

__global__ void pack_misc_kernel(const float* __restrict__ A_logs, float* __restrict__ Aexp,
                                 const float* g1, const float* b1, const float* m1, const float* v1,
                                 const float* g2, const float* b2, const float* m2, const float* v2,
                                 float* __restrict__ bn) {
  int idx = blockIdx.x * blockDim.x + threadIdx.x;
  if (idx < 16384) Aexp[idx] = -__expf(A_logs[idx]);
  if (idx < 128) {
    float s1 = g1[idx] * rsqrtf(v1[idx] + 1e-5f);
    bn[idx] = s1; bn[128 + idx] = b1[idx] - m1[idx] * s1;
    float s2 = g2[idx] * rsqrtf(v2[idx] + 1e-5f);
    bn[256 + idx] = s2; bn[384 + idx] = b2[idx] - m2[idx] * s2;
  }
}

// interleave x1/x2 into token-major f16 A matrix (M x 128)
__global__ void pack_ct_kernel(const float* __restrict__ x1, const float* __restrict__ x2,
                               _Float16* __restrict__ A) {
  int c = threadIdx.x;          // 128
  int m = blockIdx.x;           // 32768
  int b = m >> 13, l = m & 8191, y = l >> 7, wq = l & 127;
  int j = wq >> 1, par = wq & 1;
  const float* s = par ? x2 : x1;
  A[(size_t)m * 128 + c] = (_Float16)s[((size_t)(b * 128 + c) * 64 + y) * 64 + j];
}

// ------------------------------- generic GEMM ------------------------------
// 4 waves/block; one M-tile (16 rows) per wave; shared B panel staged in LDS.
// modes: 0 +bias->f32 | 1 split: n<256 raw f32, n>=256 SiLU f32 | 2 plain f32
//        3 +res f32   | 4 +bias,GELU->f16 | 5 +bias+res->f32
template <int NTT>
__global__ void gemm_kernel(const _Float16* __restrict__ A, const _Float16* __restrict__ Bp,
                            int Kd, int nTiles, int mode,
                            const float* __restrict__ bias, const float* __restrict__ res,
                            float* __restrict__ out0, float* __restrict__ out1,
                            _Float16* __restrict__ outh) {
  extern __shared__ char smem[];
  int tid = threadIdx.x;           // 0..127
  int wid = tid >> 5, lane = tid & 31;
  int lr = lane & 15, hs = lane >> 4;
  int nt0 = blockIdx.y * NTT;
  int kChunks = Kd >> 5;

  // cooperative async staging of the B panel: kChunks x (NTT KB)
  {
    const int unitsPerKc = NTT * 64;          // 16-byte units per k-chunk group
    int total = kChunks * unitsPerKc;
    for (int i = tid; i < total; i += 128) {
      int kc = i / unitsPerKc;
      int r  = i % unitsPerKc;
      const char* g = (const char*)(Bp + ((size_t)kc * nTiles + nt0) * 512) + r * 16;
      char* l = smem + ((size_t)kc * unitsPerKc + r) * 16;
      stage16(g, l);
    }
    stage_wait();
    __syncthreads();
  }

  int m0 = (blockIdx.x * 4 + wid) * 16;
  const _Float16* Arow = A + (size_t)(m0 + lr) * Kd;
  v8f c[NTT] = {};
  v16h a = load_a_frag(Arow, 0, hs);
  for (int kc = 0; kc < kChunks; ++kc) {
    v16h acur = a;
    if (kc + 1 < kChunks) {
      __builtin_prefetch(Arow + ((kc + 2) << 5), 0, 3);
      a = load_a_frag(Arow, (kc + 1) << 5, hs);
    }
#pragma unroll
    for (int j = 0; j < NTT; ++j) {
      const v16h* bp = (const v16h*)(smem + (((size_t)kc * NTT + j) << 10) + (lane << 5));
      c[j] = wmma16(acur, *bp, c[j]);
    }
  }

  int Nd = nTiles << 4;
#pragma unroll
  for (int j = 0; j < NTT; ++j) {
    int och = ((nt0 + j) << 4) + lr;
#pragma unroll
    for (int i = 0; i < 8; ++i) {
      int tok = m0 + i + hs * 8;
      float v = c[j][i];
      if (mode == 0) {
        out0[(size_t)tok * Nd + och] = v + bias[och];
      } else if (mode == 1) {
        if (och < 256) out0[(size_t)tok * 256 + och] = v;
        else           out1[(size_t)tok * 256 + (och - 256)] = v / (1.f + __expf(-v));
      } else if (mode == 2) {
        out0[(size_t)tok * Nd + och] = v;
      } else if (mode == 3) {
        out0[(size_t)tok * Nd + och] = v + res[(size_t)tok * Nd + och];
      } else if (mode == 4) {
        float t = v + bias[och];
        outh[(size_t)tok * Nd + och] = (_Float16)(0.5f * t * (1.f + erff(t * 0.70710678118f)));
      } else { // 5
        out0[(size_t)tok * Nd + och] = v + bias[och] + res[(size_t)tok * Nd + och];
      }
    }
  }
}

// ------------------ implicit-GEMM 3x3 SAME conv (128->128) -----------------
// mode 6: BN + ReLU -> f16 token-major ; mode 7: BN + residual + ReLU -> NCHW f32 (d_out)
template <int NTT>
__global__ void conv3_gemm_kernel(const _Float16* __restrict__ inAll, const _Float16* __restrict__ Bp,
                                  const float* __restrict__ bnS, const float* __restrict__ bnT,
                                  const float* __restrict__ resAll, float* __restrict__ outAll,
                                  _Float16* __restrict__ outhAll, int mode) {
  int img = blockIdx.z;
  const _Float16* A = inAll + (size_t)img * MIMG * 128;
  int lane = threadIdx.x, lr = lane & 15, hs = lane >> 4;
  int m0 = blockIdx.x * 16;
  int nt0 = blockIdx.y * NTT;
  int tok = m0 + lr;
  int bb = tok >> 12, py = (tok >> 6) & 63, px = tok & 63;
  v8f c[NTT] = {};
  const int nTiles = 8;  // 128 cols
  for (int tap = 0; tap < 9; ++tap) {
    int dy = tap / 3 - 1, dx = tap % 3 - 1;
    int y2 = py + dy, x2 = px + dx;
    bool inb = (y2 >= 0) && (y2 < 64) && (x2 >= 0) && (x2 < 64);
    const _Float16* Arow = A + (size_t)(((bb << 6) + y2) * 64 + x2) * 128;
#pragma unroll
    for (int cc = 0; cc < 4; ++cc) {
      v16h a = {};
      if (inb) a = load_a_frag(Arow, cc << 5, hs);
      int kc = tap * 4 + cc;
#pragma unroll
      for (int j = 0; j < NTT; ++j) {
        const v16h* bp = (const v16h*)(Bp + (((size_t)kc * nTiles + nt0 + j) * 32 + lane) * 16);
        c[j] = wmma16(a, *bp, c[j]);
      }
    }
  }
#pragma unroll
  for (int j = 0; j < NTT; ++j) {
    int och = ((nt0 + j) << 4) + lr;
#pragma unroll
    for (int i = 0; i < 8; ++i) {
      int t2 = m0 + i + hs * 8;
      float v = c[j][i] * bnS[och] + bnT[och];
      if (mode == 6) {
        v = fmaxf(v, 0.f);
        outhAll[((size_t)img * MIMG + t2) * 128 + och] = (_Float16)v;
      } else {
        v += resAll[((size_t)img * MIMG + t2) * 128 + och];
        v = fmaxf(v, 0.f);
        int ob = t2 >> 12, oy = (t2 >> 6) & 63, ox = t2 & 63;
        outAll[(((size_t)img * 4 + ob) * 128 + och) * 4096 + (oy << 6) + ox] = v;
      }
    }
  }
}

// ------------------------------- LayerNorm ---------------------------------
template <int C>
__global__ void ln_kernel(const float* __restrict__ x, const float* __restrict__ g,
                          const float* __restrict__ b, _Float16* __restrict__ outh) {
  int wid = threadIdx.x >> 5, lane = threadIdx.x & 31;
  int tok = blockIdx.x * 8 + wid;
  const float* xr = x + (size_t)tok * C;
  const int per = C / 32;
  float vals[per];
  float s = 0.f, sq = 0.f;
#pragma unroll
  for (int j = 0; j < per; ++j) { float v = xr[lane + (j << 5)]; vals[j] = v; s += v; sq += v * v; }
#pragma unroll
  for (int off = 16; off; off >>= 1) { s += __shfl_xor(s, off, 32); sq += __shfl_xor(sq, off, 32); }
  float mean = s / C;
  float inv = rsqrtf(sq / C - mean * mean + 1e-5f);
#pragma unroll
  for (int j = 0; j < per; ++j) {
    int idx = lane + (j << 5);
    outh[(size_t)tok * C + idx] = (_Float16)((vals[j] - mean) * inv * g[idx] + b[idx]);
  }
}

// out-norm over sum of 4 direction buffers, then gate with silu(z)
__global__ void outnorm_gate_kernel(const _Float16* __restrict__ ys, const float* __restrict__ g,
                                    const float* __restrict__ b, const float* __restrict__ zs,
                                    _Float16* __restrict__ outh) {
  int wid = threadIdx.x >> 5, lane = threadIdx.x & 31;
  int tok = blockIdx.x * 8 + wid;
  const size_t S = (size_t)MTOK * 256;
  float vals[8];
  float s = 0.f, sq = 0.f;
#pragma unroll
  for (int j = 0; j < 8; ++j) {
    size_t idx = (size_t)tok * 256 + lane + (j << 5);
    float v = (float)ys[idx] + (float)ys[S + idx] + (float)ys[2 * S + idx] + (float)ys[3 * S + idx];
    vals[j] = v; s += v; sq += v * v;
  }
#pragma unroll
  for (int off = 16; off; off >>= 1) { s += __shfl_xor(s, off, 32); sq += __shfl_xor(sq, off, 32); }
  float mean = s / 256.f;
  float inv = rsqrtf(sq / 256.f - mean * mean + 1e-5f);
#pragma unroll
  for (int j = 0; j < 8; ++j) {
    int idx = lane + (j << 5);
    float v = (vals[j] - mean) * inv * g[idx] + b[idx];
    v *= zs[(size_t)tok * 256 + idx];
    outh[(size_t)tok * 256 + idx] = (_Float16)v;
  }
}

// ------------------------ depthwise 3x3 + bias + SiLU ----------------------
__global__ void dwconv_kernel(const float* __restrict__ xp, const float* __restrict__ w,
                              const float* __restrict__ bias, _Float16* __restrict__ outh) {
  int ch = threadIdx.x;  // 256
  int m = blockIdx.x;    // 32768
  int b = m >> 13, l = m & 8191, y = l >> 7, xq = l & 127;
  float acc = bias[ch];
#pragma unroll
  for (int kh = 0; kh < 3; ++kh)
#pragma unroll
    for (int kw = 0; kw < 3; ++kw) {
      int yy = y + kh - 1, xx = xq + kw - 1;
      if (yy >= 0 && yy < 64 && xx >= 0 && xx < 128)
        acc += xp[(size_t)(((b << 6) + yy) * 128 + xx) * 256 + ch] * w[ch * 9 + kh * 3 + kw];
    }
  outh[(size_t)m * 256 + ch] = (_Float16)(acc / (1.f + __expf(-acc)));
}

// ------------------------------ selective scan -----------------------------
// one wave per (b,k,dgroup of 32); each lane owns one d-channel, 16-state recurrence.
__global__ void scan_kernel(const _Float16* __restrict__ u, const float* __restrict__ xdbl,
                            const float* __restrict__ dtw, const float* __restrict__ dtb,
                            const float* __restrict__ Aexp, const float* __restrict__ Ds,
                            _Float16* __restrict__ ysd) {
  int blk = blockIdx.x;              // 0..127
  int dgrp = blk & 7;
  int k = (blk >> 3) & 3;
  int b = blk >> 5;
  int d = dgrp * 32 + threadIdx.x;
  int kd = k * 256 + d;
  float Ar[16], w[8];
#pragma unroll
  for (int n = 0; n < 16; ++n) Ar[n] = Aexp[kd * 16 + n];
#pragma unroll
  for (int r = 0; r < 8; ++r) w[r] = dtw[kd * 8 + r];
  float bdt = dtb[kd], Dv = Ds[kd];
  float h[16];
#pragma unroll
  for (int n = 0; n < 16; ++n) h[n] = 0.f;
  const size_t S = (size_t)MTOK * 256;
  _Float16* yout = ysd + (size_t)k * S;
  for (int t = 0; t < 8192; ++t) {
    int l;
    if (k == 0)      l = t;
    else if (k == 1) { int xq = t >> 6, py = t & 63; l = (py << 7) + xq; }
    else if (k == 2) l = 8191 - t;
    else             { int tt = 8191 - t; int xq = tt >> 6, py = tt & 63; l = (py << 7) + xq; }
    int m = (b << 13) + l;
    const float* xd = xdbl + (size_t)m * 160 + k * 40;
    float uu = (float)u[(size_t)m * 256 + d];
    float dt = bdt;
#pragma unroll
    for (int r = 0; r < 8; ++r) dt += w[r] * xd[r];
    dt = (dt > 20.f) ? dt : log1pf(__expf(dt));   // softplus
    float du = dt * uu;
    float acc = 0.f;
#pragma unroll
    for (int n = 0; n < 16; ++n) {
      float a = __expf(dt * Ar[n]);
      h[n] = a * h[n] + du * xd[8 + n];
      acc += h[n] * xd[24 + n];
    }
    yout[(size_t)m * 256 + d] = (_Float16)(acc + Dv * uu);
  }
}

// de-interleave final tokens into two images (f32 residual + f16 conv input)
__global__ void deint_kernel(const float* __restrict__ xfin, float* __restrict__ xi32,
                             _Float16* __restrict__ xi16) {
  int c = threadIdx.x;    // 128
  int t = blockIdx.x;     // 0..32767
  int img = t >> 14, tok = t & 16383;
  int b = tok >> 12, y = (tok >> 6) & 63, x = tok & 63;
  int m = (b << 13) + (y << 7) + (x << 1) + img;
  float v = xfin[(size_t)m * 128 + c];
  size_t dst = ((size_t)img * MIMG + tok) * 128 + c;
  xi32[dst] = v;
  xi16[dst] = (_Float16)v;
}

static inline size_t align256(size_t x) { return (x + 255) & ~(size_t)255; }

extern "C" void kernel_launch(void* const* d_in, const int* in_sizes, int n_in,
                              void* d_out, int out_size, void* d_ws, size_t ws_size,
                              hipStream_t stream) {
  (void)in_sizes; (void)n_in; (void)out_size; (void)ws_size;
  const float* x1        = (const float*)d_in[0];
  const float* x2        = (const float*)d_in[1];
  const float* conv_in_w = (const float*)d_in[2];
  const float* conv_in_b = (const float*)d_in[3];
  const float* ln1_g     = (const float*)d_in[4];
  const float* ln1_b     = (const float*)d_in[5];
  const float* in_proj_w = (const float*)d_in[6];
  const float* dw_w      = (const float*)d_in[7];
  const float* dw_b      = (const float*)d_in[8];
  const float* x_proj_w  = (const float*)d_in[9];
  const float* dt_proj_w = (const float*)d_in[10];
  const float* dt_proj_b = (const float*)d_in[11];
  const float* A_logs    = (const float*)d_in[12];
  const float* Ds        = (const float*)d_in[13];
  const float* onorm_g   = (const float*)d_in[14];
  const float* onorm_b   = (const float*)d_in[15];
  const float* out_proj_w= (const float*)d_in[16];
  const float* ln2_g     = (const float*)d_in[17];
  const float* ln2_b     = (const float*)d_in[18];
  const float* mlp_w1    = (const float*)d_in[19];
  const float* mlp_b1    = (const float*)d_in[20];
  const float* mlp_w2    = (const float*)d_in[21];
  const float* mlp_b2    = (const float*)d_in[22];
  const float* rb1_w     = (const float*)d_in[23];
  const float* bn1_g     = (const float*)d_in[24];
  const float* bn1_b     = (const float*)d_in[25];
  const float* bn1_m     = (const float*)d_in[26];
  const float* bn1_v     = (const float*)d_in[27];
  const float* rb2_w     = (const float*)d_in[28];
  const float* bn2_g     = (const float*)d_in[29];
  const float* bn2_b     = (const float*)d_in[30];
  const float* bn2_m     = (const float*)d_in[31];
  const float* bn2_v     = (const float*)d_in[32];

  char* ws = (char*)d_ws;
  size_t off = 0;
  auto alloc = [&](size_t bytes) -> char* { char* p = ws + off; off = align256(off + bytes); return p; };

  const size_t M = MTOK;
  _Float16* Act   = (_Float16*)alloc(M * 128 * 2);
  float*    xres  = (float*)   alloc(M * 128 * 4);
  _Float16* lnx   = (_Float16*)alloc(M * 128 * 2);
  float*    xpraw = (float*)   alloc(M * 256 * 4);
  float*    zsb   = (float*)   alloc(M * 256 * 4);
  _Float16* xpf16 = (_Float16*)alloc(M * 256 * 2);
  float*    xdbl  = (float*)   alloc(M * 160 * 4);
  _Float16* ysd   = (_Float16*)alloc(4 * M * 256 * 2);
  _Float16* yg    = (_Float16*)alloc(M * 256 * 2);
  float*    xa    = (float*)   alloc(M * 128 * 4);
  _Float16* h2    = (_Float16*)alloc(M * 128 * 2);
  _Float16* mh    = (_Float16*)alloc(M * 512 * 2);
  float*    xfin  = (float*)   alloc(M * 128 * 4);
  float*    xi32  = (float*)   alloc((size_t)2 * MIMG * 128 * 4);
  _Float16* xi16  = (_Float16*)alloc((size_t)2 * MIMG * 128 * 2);
  _Float16* a1    = (_Float16*)alloc((size_t)2 * MIMG * 128 * 2);
  _Float16* Wc    = (_Float16*)alloc(128 * 128 * 2);
  _Float16* Win   = (_Float16*)alloc(128 * 512 * 2);
  _Float16* Wxp   = (_Float16*)alloc(256 * 160 * 2);
  _Float16* Wout  = (_Float16*)alloc(256 * 128 * 2);
  _Float16* Wm1   = (_Float16*)alloc(128 * 512 * 2);
  _Float16* Wm2   = (_Float16*)alloc(512 * 128 * 2);
  _Float16* Wr1   = (_Float16*)alloc(1152 * 128 * 2);
  _Float16* Wr2   = (_Float16*)alloc(1152 * 128 * 2);
  float*    Aexp  = (float*)   alloc(16384 * 4);
  float*    bnbuf = (float*)   alloc(512 * 4);

  // ---- weight packing ----
  pack_b_kernel<<<64,  256, 0, stream>>>(conv_in_w, Wc,   128, 128, 0);
  pack_b_kernel<<<256, 256, 0, stream>>>(in_proj_w, Win,  128, 512, 0);
  pack_b_kernel<<<160, 256, 0, stream>>>(x_proj_w,  Wxp,  256, 160, 0);
  pack_b_kernel<<<128, 256, 0, stream>>>(out_proj_w,Wout, 256, 128, 0);
  pack_b_kernel<<<256, 256, 0, stream>>>(mlp_w1,    Wm1,  128, 512, 0);
  pack_b_kernel<<<256, 256, 0, stream>>>(mlp_w2,    Wm2,  512, 128, 0);
  pack_b_kernel<<<576, 256, 0, stream>>>(rb1_w,     Wr1, 1152, 128, 1);
  pack_b_kernel<<<576, 256, 0, stream>>>(rb2_w,     Wr2, 1152, 128, 1);
  pack_misc_kernel<<<64, 256, 0, stream>>>(A_logs, Aexp, bn1_g, bn1_b, bn1_m, bn1_v,
                                           bn2_g, bn2_b, bn2_m, bn2_v, bnbuf);

  // ---- conv_in (1x1) as GEMM, +bias -> xres (token-major NHWC) ----
  pack_ct_kernel<<<MTOK, 128, 0, stream>>>(x1, x2, Act);
  gemm_kernel<4><<<dim3(512, 2), 128, 4 * 4 * 1024, stream>>>(
      Act, Wc, 128, 8, 0, conv_in_b, nullptr, xres, nullptr, nullptr);
  // ---- LN1 -> f16 ----
  ln_kernel<128><<<4096, 256, 0, stream>>>(xres, ln1_g, ln1_b, lnx);
  // ---- in_proj: xp raw f32 + silu(z) f32 ----
  gemm_kernel<4><<<dim3(512, 8), 128, 4 * 4 * 1024, stream>>>(
      lnx, Win, 128, 32, 1, nullptr, nullptr, xpraw, zsb, nullptr);
  // ---- depthwise 3x3 + bias + SiLU -> f16 ----
  dwconv_kernel<<<MTOK, 256, 0, stream>>>(xpraw, dw_w, dw_b, xpf16);
  // ---- x_proj for all 4 directions at once (per physical token) ----
  gemm_kernel<2><<<dim3(512, 5), 128, 8 * 2 * 1024, stream>>>(
      xpf16, Wxp, 256, 10, 2, nullptr, nullptr, xdbl, nullptr, nullptr);
  // ---- selective scan (4 directions, per-direction output buffers) ----
  scan_kernel<<<128, 32, 0, stream>>>(xpf16, xdbl, dt_proj_w, dt_proj_b, Aexp, Ds, ysd);
  // ---- out-norm over summed directions, gate with silu(z) ----
  outnorm_gate_kernel<<<4096, 256, 0, stream>>>(ysd, onorm_g, onorm_b, zsb, yg);
  // ---- out_proj + residual -> xa ----
  gemm_kernel<4><<<dim3(512, 2), 128, 8 * 4 * 1024, stream>>>(
      yg, Wout, 256, 8, 3, nullptr, xres, xa, nullptr, nullptr);
  // ---- LN2 -> f16, MLP (GELU), + residual -> xfin ----
  ln_kernel<128><<<4096, 256, 0, stream>>>(xa, ln2_g, ln2_b, h2);
  gemm_kernel<4><<<dim3(512, 8), 128, 4 * 4 * 1024, stream>>>(
      h2, Wm1, 128, 32, 4, mlp_b1, nullptr, nullptr, nullptr, mh);
  gemm_kernel<4><<<dim3(512, 2), 128, 16 * 4 * 1024, stream>>>(
      mh, Wm2, 512, 8, 5, mlp_b2, xa, xfin, nullptr, nullptr);
  // ---- de-interleave into two images ----
  deint_kernel<<<MTOK, 128, 0, stream>>>(xfin, xi32, xi16);
  // ---- resblocks: conv1(BN,ReLU) then conv2(BN,+res,ReLU) -> d_out (NCHW) ----
  conv3_gemm_kernel<4><<<dim3(1024, 2, 2), 32, 0, stream>>>(
      xi16, Wr1, bnbuf, bnbuf + 128, nullptr, nullptr, a1, 6);
  conv3_gemm_kernel<4><<<dim3(1024, 2, 2), 32, 0, stream>>>(
      a1, Wr2, bnbuf + 256, bnbuf + 384, xi32, (float*)d_out, nullptr, 7);
}
